// MutualInformationFromOtherRepo_9225589752150
// MI455X (gfx1250) — matrix operand order repaired and checked by reading, not compiled, not run on hardware
//
#include <hip/hip_runtime.h>

typedef __attribute__((ext_vector_type(16))) _Float16 v16h;
typedef __attribute__((ext_vector_type(8)))  float    v8f;

#define NBINS   256
#define NSAMP   65536     // H*W = 256*256
#define NBATCH  4
#define CHUNK   1024      // samples per workgroup
#define KBLK    32        // WMMA K per step (f16 16x16x32)

// ---------------------------------------------------------------- zero scratch
__global__ void mi_zero_kernel(float* __restrict__ p, int n) {
    int i = blockIdx.x * blockDim.x + threadIdx.x;
    if (i < n) p[i] = 0.0f;
}

// ---------------------------------------------------------------- joint + marginals
// 512 threads = 16 waves; wave w owns output row-tile (bins w*16..w*16+15).
__global__ __launch_bounds__(512)
void mi_joint_kernel(const float* __restrict__ in1, const float* __restrict__ in2,
                     float* __restrict__ joint, float* __restrict__ marg1,
                     float* __restrict__ marg2)
{
    __shared__ float sx1[KBLK];
    __shared__ float sx2[KBLK];
    __shared__ __align__(32) _Float16 kv2[NBINS][KBLK];   // 16 KB f16 tile

    const int b      = blockIdx.y;
    const int chunk0 = blockIdx.x * CHUNK;
    const int tid    = threadIdx.x;
    const int lane   = tid & 31;
    const int wave   = tid >> 5;          // 0..15 == output row tile index
    const int mrow   = lane & 15;         // M (and B-column) within tile
    const int khalf  = (lane >> 4) * 16;  // which K-half this lane holds

    const float* x1 = in1 + (size_t)b * NSAMP;
    const float* x2 = in2 + (size_t)b * NSAMP;

    v8f acc[16];
    const v8f vzero = {0.f,0.f,0.f,0.f,0.f,0.f,0.f,0.f};
#pragma unroll
    for (int j = 0; j < 16; ++j) acc[j] = vzero;

    float s1 = 0.0f;                      // marginal accum for bin wave*16+mrow
    float s2 = 0.0f;                      // marginal accum for fill-role bin
    const int fbin = tid >> 1;            // kv2 fill: one bin per thread pair
    const int fkh  = (tid & 1) * 16;

    for (int kb = 0; kb < CHUNK; kb += KBLK) {
        __syncthreads();                  // previous tile fully consumed
        if (tid < KBLK)             sx1[tid]        = x1[chunk0 + kb + tid]        * 255.0f;
        else if (tid < 2 * KBLK)    sx2[tid - KBLK] = x2[chunk0 + kb + tid - KBLK] * 255.0f;
        __syncthreads();

        // --- cooperatively build kv2 tile (256 bins x 32 samples), each exp once
        {
            const float bin = (float)fbin;
#pragma unroll
            for (int kk = 0; kk < 16; ++kk) {
                float d = sx2[fkh + kk] - bin;
                float v = __expf(-50.0f * d * d);
                kv2[fbin][fkh + kk] = (_Float16)v;
                s2 += v;
            }
        }
        __syncthreads();

        // --- build this wave's A fragment (contiguous k = 16*(lane>>4) + s, same k order as B)
        union { _Float16 h[16]; v16h v; } afrag;
        {
            const float bin = (float)(wave * 16 + mrow);
#pragma unroll
            for (int r = 0; r < 8; ++r) {
#pragma unroll
                for (int hh = 0; hh < 2; ++hh) {
                    int k = ((lane >> 4) * 16) + r * 2 + hh;
                    float d = sx1[k] - bin;
                    float e = __expf(-50.0f * d * d);
                    afrag.h[2 * r + hh] = (_Float16)e;
                    s1 += e;
                }
            }
        }

        // --- 16 WMMAs: this row-tile against every column tile of kv2
#pragma unroll
        for (int j = 0; j < 16; ++j) {
            v16h bfrag = *(const v16h*)&kv2[j * 16 + mrow][khalf];
            acc[j] = __builtin_amdgcn_wmma_f32_16x16x32_f16(
                         false, afrag.v, false, bfrag,
                         (short)0, acc[j], false, false);
        }
    }

    // --- flush joint tiles (C/D layout: VGPR r -> M = r + 8*(lane>>4), N = lane&15)
    float* jb = joint + (size_t)b * NBINS * NBINS;
#pragma unroll
    for (int j = 0; j < 16; ++j) {
#pragma unroll
        for (int r = 0; r < 8; ++r) {
            int row = wave * 16 + r + (lane >> 4) * 8;
            int col = j * 16 + (lane & 15);
            atomicAdd(&jb[row * NBINS + col], acc[j][r]);
        }
    }
    atomicAdd(&marg1[b * NBINS + wave * 16 + mrow], s1);
    atomicAdd(&marg2[b * NBINS + fbin], s2);
}

// ---------------------------------------------------------------- entropies + NMI
__device__ __forceinline__ float mi_block_reduce(float v, float* red, int t) {
    red[t] = v; __syncthreads();
    for (int s = 128; s > 0; s >>= 1) {
        if (t < s) red[t] += red[t + s];
        __syncthreads();
    }
    float r = red[0]; __syncthreads();
    return r;
}

__global__ __launch_bounds__(256)
void mi_finalize_kernel(const float* __restrict__ joint,
                        const float* __restrict__ marg1,
                        const float* __restrict__ marg2,
                        float* __restrict__ out)
{
    __shared__ float red[256];
    const int t = threadIdx.x;
    const float EPSF = 1e-10f;
    const float invN = 1.0f / (float)NSAMP;
    float acc_nmi = 0.0f;

    for (int b = 0; b < NBATCH; ++b) {
        // H(x1)
        float m1 = marg1[b * NBINS + t] * invN;
        float sum1 = mi_block_reduce(m1, red, t);
        float p1 = m1 / (sum1 + EPSF);
        float H1 = mi_block_reduce(-p1 * __log2f(p1 + EPSF), red, t);

        // H(x2)
        float m2 = marg2[b * NBINS + t] * invN;
        float sum2 = mi_block_reduce(m2, red, t);
        float p2 = m2 / (sum2 + EPSF);
        float H2 = mi_block_reduce(-p2 * __log2f(p2 + EPSF), red, t);

        // H(x1,x2)
        const float* jb = joint + (size_t)b * NBINS * NBINS;
        float ls = 0.0f;
        for (int i = t; i < NBINS * NBINS; i += 256) ls += jb[i];
        float sj = mi_block_reduce(ls, red, t);
        float inv = 1.0f / (sj + EPSF);
        float lh = 0.0f;
        for (int i = t; i < NBINS * NBINS; i += 256) {
            float p = jb[i] * inv;
            lh -= p * __log2f(p + EPSF);
        }
        float Hj = mi_block_reduce(lh, red, t);

        float mi = H1 + H2 - Hj;
        acc_nmi += 2.0f * mi / (H1 + H2);
    }
    if (t == 0) out[0] = acc_nmi / (float)NBATCH;
}

// ---------------------------------------------------------------- launch
extern "C" void kernel_launch(void* const* d_in, const int* in_sizes, int n_in,
                              void* d_out, int out_size, void* d_ws, size_t ws_size,
                              hipStream_t stream) {
    const float* in1 = (const float*)d_in[0];
    const float* in2 = (const float*)d_in[1];

    float* joint = (float*)d_ws;                              // B*256*256 f32 = 1 MB
    float* marg1 = joint + (size_t)NBATCH * NBINS * NBINS;    // B*256 f32
    float* marg2 = marg1 + (size_t)NBATCH * NBINS;            // B*256 f32
    float* out   = (float*)d_out;

    const int nzero = NBATCH * NBINS * NBINS + 2 * NBATCH * NBINS;
    mi_zero_kernel<<<(nzero + 255) / 256, 256, 0, stream>>>(joint, nzero);

    dim3 grid(NSAMP / CHUNK, NBATCH);   // 64 x 4 workgroups
    mi_joint_kernel<<<grid, 512, 0, stream>>>(in1, in2, joint, marg1, marg2);

    mi_finalize_kernel<<<1, 256, 0, stream>>>(joint, marg1, marg2, out);
}
